// EncoderLayer_28965259444902
// MI455X (gfx1250) — compile-verified, matrix operand order checked by reading, not run on hardware
//
#include <hip/hip_runtime.h>

typedef _Float16 h16;
typedef __attribute__((ext_vector_type(16))) _Float16 v16h;
typedef __attribute__((ext_vector_type(8)))  float    v8f;

#define D_MODEL 512
#define D_INNER 1024
#define D_STATE 16
#define DT_RANK 32
#define D_CONV  4
#define REG_SZ  256
#define NREG    64
#define BATCH   2
#define ROWS    (BATCH * NREG * REG_SZ)      /* 32768 */
#define XDBC_W  (DT_RANK + 2 * D_STATE)      /* 64 */
#define ANG_STEP 8.97612087208418e-5f        /* 2*pi / (70000-1) */

// ---------------------------------------------------------------------------
// Elementwise helpers
// ---------------------------------------------------------------------------
__global__ void f32_to_f16_kernel(const float* __restrict__ src,
                                  h16* __restrict__ dst, int n) {
  int i = blockIdx.x * blockDim.x + threadIdx.x;
  if (i < n) dst[i] = (h16)src[i];
}

__global__ void rotary_cast_kernel(const float* __restrict__ x,
                                   const float* __restrict__ mask,
                                   h16* __restrict__ out) {
  int idx = blockIdx.x * blockDim.x + threadIdx.x;
  if (idx >= ROWS * D_MODEL) return;
  int row = idx / D_MODEL;
  int d   = idx - row * D_MODEL;
  int t   = row & (REG_SZ - 1);
  float ang = (float)t * ANG_STEP;
  float c = __cosf(ang), s = __sinf(ang);
  int dm1 = (d == 0) ? (D_MODEL - 1) : (d - 1);
  float v = x[(size_t)row * D_MODEL + d] * c
          + x[(size_t)row * D_MODEL + dm1] * s;
  out[idx] = (h16)(v * mask[row]);
}

__global__ void conv_silu_kernel(const h16* __restrict__ u,
                                 const float* __restrict__ cw,
                                 const float* __restrict__ cb,
                                 h16* __restrict__ out) {
  int idx = blockIdx.x * blockDim.x + threadIdx.x;
  if (idx >= ROWS * D_INNER) return;
  int row = idx / D_INNER;
  int d   = idx - row * D_INNER;
  int t   = row & (REG_SZ - 1);
  float acc = cb[d];
#pragma unroll
  for (int j = 0; j < D_CONV; ++j) {
    int tt = t - (D_CONV - 1) + j;
    if (tt >= 0)
      acc += (float)u[(size_t)(row - (D_CONV - 1) + j) * D_INNER + d]
             * cw[d * D_CONV + j];
  }
  float sig = 1.0f / (1.0f + __expf(-acc));
  out[idx] = (h16)(acc * sig);
}

__global__ void slice_dtin_kernel(const float* __restrict__ xdbc,
                                  h16* __restrict__ dst) {
  int idx = blockIdx.x * blockDim.x + threadIdx.x;
  if (idx >= ROWS * DT_RANK) return;
  int row = idx / DT_RANK;
  int j   = idx - row * DT_RANK;
  dst[idx] = (h16)xdbc[(size_t)row * XDBC_W + j];
}

__global__ void copy_mask_kernel(const float* __restrict__ mask,
                                 float* __restrict__ dst) {
  int i = blockIdx.x * blockDim.x + threadIdx.x;
  if (i < ROWS) dst[i] = mask[i];
}

// ---------------------------------------------------------------------------
// Double-buffered WMMA GEMM: C[MxN] = A[MxK] * B[KxN], f16 in / f32 acc.
// Block tile BM=64 x BN (64 or 128), BK=32, 8 waves (256 threads).
// Wave (wm,wn) computes 16 x (BN/2) via NT=BN/32 v_wmma_f32_16x16x32_f16.
// Tile k+1 is prefetched into registers while tile k feeds the WMMAs;
// global_prefetch covers tile k+2.
// MODE 0: f32 to Cf.  MODE 1: split u/z f16 to Ch/Ch2.
// MODE 2: softplus(acc + bias[col]) f16 to Ch.
// ---------------------------------------------------------------------------
template <int MODE, int BN>
__global__ __launch_bounds__(256)
void gemm_wmma_kernel(const h16* __restrict__ A, const h16* __restrict__ B,
                      int M, int N, int K,
                      float* __restrict__ Cf, h16* __restrict__ Ch,
                      h16* __restrict__ Ch2, const float* __restrict__ bias) {
  constexpr int BM = 64, BK = 32, PAD = 8;
  constexpr int NT  = BN / 32;   // wmma col-tiles per wave
  constexpr int BCH = BN / 64;   // 16B B-chunks per thread per tile
  __shared__ h16 As[2][BM][BK + PAD];
  __shared__ h16 Bs[2][BN][BK + PAD];   // transposed: Bs[buf][n][k]

  const int tid  = threadIdx.x;
  const int lane = tid & 31;
  const int w    = tid >> 5;
  const int wm   = w & 3;    // 16-row strip
  const int wn   = w >> 2;   // (BN/2)-col strip
  const int bM   = blockIdx.y * BM;
  const int bN   = blockIdx.x * BN;
  const int ml   = lane & 15;
  const int kb   = (lane < 16) ? 0 : 8;

  const int ar = tid >> 2;          // A chunk row 0..63
  const int ac = (tid & 3) * 8;     // A chunk col 0,8,16,24

  v8f acc[NT];
  const v8f vzero = {0.f, 0.f, 0.f, 0.f, 0.f, 0.f, 0.f, 0.f};
#pragma unroll
  for (int i = 0; i < NT; ++i) acc[i] = vzero;

  uint4 aReg;
  uint4 bReg[BCH];

  auto loadTile = [&](int k0) {
    aReg = *(const uint4*)(A + (size_t)(bM + ar) * K + k0 + ac);
#pragma unroll
    for (int i = 0; i < BCH; ++i) {
      int q  = tid + 256 * i;
      int kk = q / (BN / 8);
      int nb = (q % (BN / 8)) * 8;
      bReg[i] = *(const uint4*)(B + (size_t)(k0 + kk) * N + bN + nb);
    }
  };
  auto storeTile = [&](int buf) {
    *(uint4*)(&As[buf][ar][ac]) = aReg;
#pragma unroll
    for (int i = 0; i < BCH; ++i) {
      int q  = tid + 256 * i;
      int kk = q / (BN / 8);
      int nb = (q % (BN / 8)) * 8;
      const h16* hv = (const h16*)&bReg[i];
#pragma unroll
      for (int j = 0; j < 8; ++j) Bs[buf][nb + j][kk] = hv[j];
    }
  };

  const int nk = K / BK;
  loadTile(0);
  storeTile(0);
  __syncthreads();

  for (int kt = 0; kt < nk; ++kt) {
    const int cur = kt & 1;
    if (kt + 1 < nk) loadTile((kt + 1) * BK);
    if (kt + 2 < nk) {
      __builtin_prefetch(A + (size_t)(bM + ar) * K + (kt + 2) * BK + ac, 0, 1);
      __builtin_prefetch(B + (size_t)((kt + 2) * BK + (tid >> 3)) * N + bN, 0, 1);
    }

    v16h a = {};
    v16h bf[NT] = {};
#pragma unroll
    for (int v = 0; v < 8; ++v) {
      int kk = (v >> 2) * 16 + kb + (v & 3) * 2;   // ISA 16-bit A/B layout
      a[2 * v]     = As[cur][wm * 16 + ml][kk];
      a[2 * v + 1] = As[cur][wm * 16 + ml][kk + 1];
#pragma unroll
      for (int nt = 0; nt < NT; ++nt) {
        int nr = wn * (BN / 2) + nt * 16 + ml;
        bf[nt][2 * v]     = Bs[cur][nr][kk];
        bf[nt][2 * v + 1] = Bs[cur][nr][kk + 1];
      }
    }
#pragma unroll
    for (int nt = 0; nt < NT; ++nt)
      acc[nt] = __builtin_amdgcn_wmma_f32_16x16x32_f16(
          false, a, false, bf[nt], (short)0, acc[nt], false, false);

    if (kt + 1 < nk) {
      storeTile(1 - cur);   // implicit s_wait_loadcnt happens here, after WMMAs
      __syncthreads();
    }
  }

  // Epilogue. C/D layout: VGPR e holds M = e + 8*(lane>=16), N = lane&15.
  const int rowBase = bM + wm * 16 + ((lane < 16) ? 0 : 8);
#pragma unroll
  for (int nt = 0; nt < NT; ++nt) {
    const int col = bN + wn * (BN / 2) + nt * 16 + ml;
#pragma unroll
    for (int e = 0; e < 8; ++e) {
      const int row = rowBase + e;
      float v = acc[nt][e];
      if constexpr (MODE == 0) {
        Cf[(size_t)row * N + col] = v;
      } else if constexpr (MODE == 1) {
        if (col < D_INNER) Ch [(size_t)row * D_INNER + col]           = (h16)v;
        else               Ch2[(size_t)row * D_INNER + col - D_INNER] = (h16)v;
      } else {  // MODE 2
        float t  = v + bias[col];
        float sp = (t > 20.f) ? t : log1pf(__expf(t));
        Ch[(size_t)row * N + col] = (h16)sp;
      }
    }
  }
}

// ---------------------------------------------------------------------------
// Selective scan: one block per region (1024 threads = one channel each).
// B_t/C_t staged in LDS (32 KB); h[16] states live in registers.
// Fused epilogue: y = (scan + u*D_skip) * silu(z), stored f16 for final GEMM.
// ---------------------------------------------------------------------------
__global__ __launch_bounds__(1024)
void scan_kernel(const float* __restrict__ xdbc, const h16* __restrict__ dt,
                 const h16* __restrict__ uc, const h16* __restrict__ z,
                 const float* __restrict__ A_log, const float* __restrict__ Dsk,
                 h16* __restrict__ y) {
  __shared__ float Bs[REG_SZ][D_STATE];
  __shared__ float Cs[REG_SZ][D_STATE];
  const int reg = blockIdx.x;
  const int d   = threadIdx.x;

  for (int i = threadIdx.x; i < REG_SZ * 2 * D_STATE; i += blockDim.x) {
    int t = i / (2 * D_STATE);
    int j = i - t * (2 * D_STATE);
    float v = xdbc[((size_t)reg * REG_SZ + t) * XDBC_W + DT_RANK + j];
    if (j < D_STATE) Bs[t][j] = v;
    else             Cs[t][j - D_STATE] = v;
  }
  __syncthreads();

  float Ad[D_STATE];
#pragma unroll
  for (int s = 0; s < D_STATE; ++s) Ad[s] = -__expf(A_log[d * D_STATE + s]);
  const float Dd = Dsk[d];

  float h[D_STATE];
#pragma unroll
  for (int s = 0; s < D_STATE; ++s) h[s] = 0.f;

  for (int t = 0; t < REG_SZ; ++t) {
    size_t o = ((size_t)reg * REG_SZ + t) * D_INNER + d;
    float dtv = (float)dt[o];
    float uv  = (float)uc[o];
    float du  = dtv * uv;
    float yv  = 0.f;
#pragma unroll
    for (int s = 0; s < D_STATE; ++s) {
      float dA = __expf(dtv * Ad[s]);
      h[s] = dA * h[s] + du * Bs[t][s];
      yv += h[s] * Cs[t][s];
    }
    float zv = (float)z[o];
    float g  = zv / (1.0f + __expf(-zv));
    y[o] = (h16)((yv + uv * Dd) * g);
  }
}

// ---------------------------------------------------------------------------
extern "C" void kernel_launch(void* const* d_in, const int* in_sizes, int n_in,
                              void* d_out, int out_size, void* d_ws,
                              size_t ws_size, hipStream_t stream) {
  const float* x      = (const float*)d_in[0];
  const float* mask   = (const float*)d_in[1];
  const float* W_in   = (const float*)d_in[3];
  const float* conv_w = (const float*)d_in[4];
  const float* conv_b = (const float*)d_in[5];
  const float* W_x    = (const float*)d_in[6];
  const float* W_dt   = (const float*)d_in[7];
  const float* b_dt   = (const float*)d_in[8];
  const float* A_log  = (const float*)d_in[9];
  const float* D_skip = (const float*)d_in[10];
  const float* W_out  = (const float*)d_in[11];
  float* out = (float*)d_out;

  char* ws = (char*)d_ws;
  size_t off = 0;
  auto take = [&](size_t bytes) -> char* {
    char* p = ws + off;
    off += (bytes + 255) & ~(size_t)255;
    return p;
  };
  h16*   aF16    = (h16*)take((size_t)ROWS * D_MODEL * 2);
  h16*   wInF16  = (h16*)take((size_t)D_MODEL * 2 * D_INNER * 2);
  h16*   wXF16   = (h16*)take((size_t)D_INNER * XDBC_W * 2);
  h16*   wDtF16  = (h16*)take((size_t)DT_RANK * D_INNER * 2);
  h16*   wOutF16 = (h16*)take((size_t)D_INNER * D_MODEL * 2);
  h16*   uF16    = (h16*)take((size_t)ROWS * D_INNER * 2);
  h16*   zF16    = (h16*)take((size_t)ROWS * D_INNER * 2);
  h16*   cF16    = (h16*)take((size_t)ROWS * D_INNER * 2);
  float* xdbcF32 = (float*)take((size_t)ROWS * XDBC_W * 4);
  h16*   dtInF16 = (h16*)take((size_t)ROWS * DT_RANK * 2);
  h16*   dtF16   = (h16*)take((size_t)ROWS * D_INNER * 2);
  h16*   yF16    = (h16*)take((size_t)ROWS * D_INNER * 2);

  const int T = 256;
  auto blk = [&](int n) { return (n + T - 1) / T; };

  // Weight conversion f32 -> f16
  f32_to_f16_kernel<<<blk(D_MODEL * 2 * D_INNER), T, 0, stream>>>(W_in, wInF16, D_MODEL * 2 * D_INNER);
  f32_to_f16_kernel<<<blk(D_INNER * XDBC_W), T, 0, stream>>>(W_x, wXF16, D_INNER * XDBC_W);
  f32_to_f16_kernel<<<blk(DT_RANK * D_INNER), T, 0, stream>>>(W_dt, wDtF16, DT_RANK * D_INNER);
  f32_to_f16_kernel<<<blk(D_INNER * D_MODEL), T, 0, stream>>>(W_out, wOutF16, D_INNER * D_MODEL);

  // Rotary + mask + cast
  rotary_cast_kernel<<<blk(ROWS * D_MODEL), T, 0, stream>>>(x, mask, aF16);

  // GEMM1: (32768x512) x (512x2048) -> split u/z halves (f16)
  {
    dim3 g(2 * D_INNER / 128, ROWS / 64);
    gemm_wmma_kernel<1, 128><<<g, 256, 0, stream>>>(aF16, wInF16, ROWS,
        2 * D_INNER, D_MODEL, nullptr, uF16, zF16, nullptr);
  }

  // Depthwise causal conv + SiLU
  conv_silu_kernel<<<blk(ROWS * D_INNER), T, 0, stream>>>(uF16, conv_w, conv_b, cF16);

  // GEMM2: (32768x1024) x (1024x64) -> xdbc f32
  {
    dim3 g(XDBC_W / 64, ROWS / 64);
    gemm_wmma_kernel<0, 64><<<g, 256, 0, stream>>>(cF16, wXF16, ROWS, XDBC_W,
        D_INNER, xdbcF32, nullptr, nullptr, nullptr);
  }

  // Slice dt-rank columns -> f16
  slice_dtin_kernel<<<blk(ROWS * DT_RANK), T, 0, stream>>>(xdbcF32, dtInF16);

  // GEMM-dt: (32768x32) x (32x1024) + b_dt -> softplus -> dt f16 (single K step)
  {
    dim3 g(D_INNER / 128, ROWS / 64);
    gemm_wmma_kernel<2, 128><<<g, 256, 0, stream>>>(dtInF16, wDtF16, ROWS,
        D_INNER, DT_RANK, nullptr, dtF16, nullptr, b_dt);
  }

  // Selective scan fused with D-skip + z-gate
  scan_kernel<<<BATCH * NREG, 1024, 0, stream>>>(xdbcF32, dtF16, cF16, zF16,
                                                 A_log, D_skip, yF16);

  // GEMM3: (32768x1024) x (1024x512) -> out f32 directly into d_out
  {
    dim3 g(D_MODEL / 128, ROWS / 64);
    gemm_wmma_kernel<0, 128><<<g, 256, 0, stream>>>(yF16, wOutF16, ROWS,
        D_MODEL, D_INNER, out, nullptr, nullptr, nullptr);
  }

  // Second tuple output: mask passthrough
  copy_mask_kernel<<<blk(ROWS), T, 0, stream>>>(mask, out + (size_t)ROWS * D_MODEL);
}
